// AttentionStyleEstimator_26362509263641
// MI455X (gfx1250) — matrix operand order repaired and checked by reading, not compiled, not run on hardware
//
#include <hip/hip_runtime.h>
#include <stdint.h>

#define DIMF 512
#define NHEADS 8
#define HDIM 64
#define NB 2
#define NSEQ 2048
#define NTOK (NB * NSEQ)      // 4096
#define KNEI 32
#define SCALEF 0.125f         // 64^-0.5
#define LSTR 2052             // padded LDS row stride (words)

typedef __attribute__((ext_vector_type(8)))  __bf16 v8bf;
typedef __attribute__((ext_vector_type(16))) __bf16 v16bf;
typedef __attribute__((ext_vector_type(8)))  float  v8f;

// ---------- helpers ----------
__device__ __forceinline__ unsigned short f2bf(float f) {
  unsigned u = __float_as_uint(f);
  unsigned r = u + 0x7FFFu + ((u >> 16) & 1u);   // RNE
  return (unsigned short)(r >> 16);
}
__device__ __forceinline__ float bf2f(unsigned short h) {
  return __uint_as_float(((unsigned)h) << 16);
}
// two contiguous 16B chunks -> one 16-element bf16 operand (VGPRs 0..3 / 4..7)
__device__ __forceinline__ v16bf ld16(const unsigned short* p0, const unsigned short* p1) {
  v8bf a = *(const v8bf*)p0;
  v8bf b = *(const v8bf*)p1;
  return __builtin_shufflevector(a, b, 0,1,2,3,4,5,6,7,8,9,10,11,12,13,14,15);
}
// order-preserving f32 <-> u32 key
__device__ __forceinline__ unsigned f2key(float s) {
  unsigned b = __float_as_uint(s);
  return (b & 0x80000000u) ? ~b : (b | 0x80000000u);
}
__device__ __forceinline__ float key2f(unsigned k) {
  unsigned b = (k & 0x80000000u) ? (k ^ 0x80000000u) : ~k;
  return __uint_as_float(b);
}
// wave32 butterfly reductions via ds_swizzle (xor-mask mode, and=0x1f)
__device__ __forceinline__ int wred_sum_i(int v) {
  v += __builtin_amdgcn_ds_swizzle(v, 0x401F);
  v += __builtin_amdgcn_ds_swizzle(v, 0x201F);
  v += __builtin_amdgcn_ds_swizzle(v, 0x101F);
  v += __builtin_amdgcn_ds_swizzle(v, 0x081F);
  v += __builtin_amdgcn_ds_swizzle(v, 0x041F);
  return v;
}
__device__ __forceinline__ unsigned wred_max_u(unsigned v) {
  unsigned o;
  o = (unsigned)__builtin_amdgcn_ds_swizzle((int)v, 0x401F); v = o > v ? o : v;
  o = (unsigned)__builtin_amdgcn_ds_swizzle((int)v, 0x201F); v = o > v ? o : v;
  o = (unsigned)__builtin_amdgcn_ds_swizzle((int)v, 0x101F); v = o > v ? o : v;
  o = (unsigned)__builtin_amdgcn_ds_swizzle((int)v, 0x081F); v = o > v ? o : v;
  o = (unsigned)__builtin_amdgcn_ds_swizzle((int)v, 0x041F); v = o > v ? o : v;
  return v;
}
__device__ __forceinline__ float wred_sum_f(float v) {
  v += __int_as_float(__builtin_amdgcn_ds_swizzle(__float_as_int(v), 0x401F));
  v += __int_as_float(__builtin_amdgcn_ds_swizzle(__float_as_int(v), 0x201F));
  v += __int_as_float(__builtin_amdgcn_ds_swizzle(__float_as_int(v), 0x101F));
  v += __int_as_float(__builtin_amdgcn_ds_swizzle(__float_as_int(v), 0x081F));
  v += __int_as_float(__builtin_amdgcn_ds_swizzle(__float_as_int(v), 0x041F));
  return v;
}

// ---------- kernel 0a: split f32 -> bf16 hi/lo ----------
__global__ void k_split(const float* __restrict__ src,
                        unsigned short* __restrict__ hi,
                        unsigned short* __restrict__ lo, int n) {
  int i = blockIdx.x * blockDim.x + threadIdx.x;
  if (i >= n) return;
  float v = src[i];
  unsigned short h = f2bf(v);
  hi[i] = h;
  lo[i] = f2bf(v - bf2f(h));
}

// ---------- kernel 0b: W[j][d] -> [j][d] (same layout, no transpose), split hi/lo ----------
__global__ void k_splitT(const float* __restrict__ W,
                         unsigned short* __restrict__ hi,
                         unsigned short* __restrict__ lo) {
  int t = blockIdx.x * blockDim.x + threadIdx.x;   // 512*512 threads
  int j = t >> 9, d = t & 511;
  float v = W[t];                                  // W[j][d], coalesced read
  unsigned short h = f2bf(v);
  int o = j * DIMF + d;
  hi[o] = h;
  lo[o] = f2bf(v - bf2f(h));
}

// ---------- kernel 1: projection q/k = x @ W^T + b (bf16x3 WMMA) ----------
// one wave per 16x16 output tile; writes Q and K both as [b,h,n,d]
__global__ void k_proj(const unsigned short* __restrict__ xhi,
                       const unsigned short* __restrict__ xlo,
                       const unsigned short* __restrict__ wthi,
                       const unsigned short* __restrict__ wtlo,
                       const float* __restrict__ bias,
                       unsigned short* __restrict__ ohi,
                       unsigned short* __restrict__ olo,
                       int is_k) {
  const int lane = threadIdx.x & 31;
  const int lm = lane & 15, lh = lane >> 4;
  const int i0 = blockIdx.x * 16;        // token tile
  const int c0 = blockIdx.y * 16;        // output-feature tile
  v8f acc = {};
  const unsigned short* xhp = xhi + (i0 + lm) * DIMF + lh * 8;
  const unsigned short* xlp = xlo + (i0 + lm) * DIMF + lh * 8;
  #pragma unroll
  for (int ks = 0; ks < 16; ++ks) {
    const int ka = ks * 32;
    v16bf ah = ld16(xhp + ka, xhp + ka + 16);
    v16bf al = ld16(xlp + ka, xlp + ka + 16);
    const unsigned short* bhp = wthi + (c0 + lm) * DIMF + ka + lh * 8;  // lane = N-col
    const unsigned short* blp = wtlo + (c0 + lm) * DIMF + ka + lh * 8;
    v16bf bh = ld16(bhp, bhp + 16);
    v16bf bl = ld16(blp, blp + 16);
    acc = __builtin_amdgcn_wmma_f32_16x16x32_bf16(false, ah, false, bh, (short)0, acc, false, false);
    acc = __builtin_amdgcn_wmma_f32_16x16x32_bf16(false, ah, false, bl, (short)0, acc, false, false);
    acc = __builtin_amdgcn_wmma_f32_16x16x32_bf16(false, al, false, bh, (short)0, acc, false, false);
  }
  const int j = c0 + lm;
  const float bj = bias[j];
  const int head = j >> 6, d = j & 63;
  #pragma unroll
  for (int r = 0; r < 8; ++r) {
    const int m = r + 8 * lh;            // D row of tile
    const int i = i0 + m;
    const int b = i >> 11, n = i & (NSEQ - 1);
    float val = acc[r] + bj;
    unsigned short h = f2bf(val);
    unsigned short l = f2bf(val - bf2f(h));
    int idx;
    if (!is_k) idx = ((b * NHEADS + head) * NSEQ + n) * HDIM + d;   // Q row-major
    else       idx = ((b * NHEADS + head) * NSEQ + n) * HDIM + d;   // K row-major
    ohi[idx] = h;
    olo[idx] = l;
  }
}

// ---------- kernel 2: fused scores + exact top-32 + masked softmax ----------
// block = 512 threads (16 waves), one (b,h) x 16-row block; LDS = 16 x 2052 f32
__global__ void k_attn(const unsigned short* __restrict__ qhi,
                       const unsigned short* __restrict__ qlo,
                       const unsigned short* __restrict__ kthi,
                       const unsigned short* __restrict__ ktlo,
                       float* __restrict__ out) {
  extern __shared__ float lds[];
  const int lane = threadIdx.x & 31;
  const int w = threadIdx.x >> 5;
  const int lm = lane & 15, lh = lane >> 4;
  const int n0 = blockIdx.x * 16;
  const int bh = blockIdx.y;

  // A operands: Q rows n0..n0+15 of this head, both 32-wide K steps, hi+lo
  const unsigned short* qhp = qhi + ((bh * NSEQ) + n0 + lm) * HDIM + lh * 8;
  const unsigned short* qlp = qlo + ((bh * NSEQ) + n0 + lm) * HDIM + lh * 8;
  v16bf ah0 = ld16(qhp,      qhp + 16);
  v16bf al0 = ld16(qlp,      qlp + 16);
  v16bf ah1 = ld16(qhp + 32, qhp + 48);
  v16bf al1 = ld16(qlp + 32, qlp + 48);

  // phase 1: each wave computes 8 of the 128 column tiles
  #pragma unroll
  for (int tt = 0; tt < 8; ++tt) {
    const int c0 = (w + 16 * tt) * 16;
    v8f acc = {};
    const unsigned short* khp = kthi + ((bh * NSEQ) + c0 + lm) * HDIM + lh * 8;  // lane = key col
    const unsigned short* klp = ktlo + ((bh * NSEQ) + c0 + lm) * HDIM + lh * 8;
    v16bf bh0 = ld16(khp, khp + 16);
    v16bf bl0 = ld16(klp, klp + 16);
    acc = __builtin_amdgcn_wmma_f32_16x16x32_bf16(false, ah0, false, bh0, (short)0, acc, false, false);
    acc = __builtin_amdgcn_wmma_f32_16x16x32_bf16(false, ah0, false, bl0, (short)0, acc, false, false);
    acc = __builtin_amdgcn_wmma_f32_16x16x32_bf16(false, al0, false, bh0, (short)0, acc, false, false);
    const unsigned short* khp1 = khp + 32;   // d = 32..63
    const unsigned short* klp1 = klp + 32;
    v16bf bh1 = ld16(khp1, khp1 + 16);
    v16bf bl1 = ld16(klp1, klp1 + 16);
    acc = __builtin_amdgcn_wmma_f32_16x16x32_bf16(false, ah1, false, bh1, (short)0, acc, false, false);
    acc = __builtin_amdgcn_wmma_f32_16x16x32_bf16(false, ah1, false, bl1, (short)0, acc, false, false);
    acc = __builtin_amdgcn_wmma_f32_16x16x32_bf16(false, al1, false, bh1, (short)0, acc, false, false);
    #pragma unroll
    for (int r = 0; r < 8; ++r) {
      const int m = r + 8 * lh;
      lds[m * LSTR + c0 + lm] = acc[r] * SCALEF;
    }
  }
  __syncthreads();

  // phase 2: wave w owns row w; lane holds cols lane+32j as sortable keys
  unsigned key[64];
  {
    const float* row = lds + w * LSTR;
    #pragma unroll
    for (int j = 0; j < 64; ++j) key[j] = f2key(row[lane + 32 * j]);
  }
  // exact radix select: largest T with count(key >= T) >= 32  == 32nd-largest key
  unsigned prefix = 0u;
  for (int bit = 31; bit >= 0; --bit) {
    const unsigned cand = prefix | (1u << bit);
    int c = 0;
    #pragma unroll
    for (int j = 0; j < 64; ++j) c += (key[j] >= cand) ? 1 : 0;
    c = wred_sum_i(c);
    if (c >= KNEI) prefix = cand;
  }
  // row max (max key is always kept)
  unsigned gk = 0u;
  #pragma unroll
  for (int j = 0; j < 64; ++j) gk = key[j] > gk ? key[j] : gk;
  gk = wred_max_u(gk);
  const float gmax = key2f(gk);
  // masked softmax denominator
  float sum = 0.0f;
  #pragma unroll
  for (int j = 0; j < 64; ++j)
    sum += (key[j] >= prefix) ? __expf(key2f(key[j]) - gmax) : 0.0f;
  sum = wred_sum_f(sum);
  const float rs = 1.0f / sum;
  float* orow = out + ((long)bh * NSEQ + n0 + w) * NSEQ;
  #pragma unroll
  for (int j = 0; j < 64; ++j) {
    float p = (key[j] >= prefix) ? __expf(key2f(key[j]) - gmax) * rs : 0.0f;
    orow[lane + 32 * j] = p;   // coalesced
  }
}

// ---------- host ----------
extern "C" void kernel_launch(void* const* d_in, const int* in_sizes, int n_in,
                              void* d_out, int out_size, void* d_ws, size_t ws_size,
                              hipStream_t stream) {
  const float* x  = (const float*)d_in[0];
  const float* WQ = (const float*)d_in[1];
  const float* bQ = (const float*)d_in[2];
  const float* WK = (const float*)d_in[3];
  const float* bK = (const float*)d_in[4];
  float* out = (float*)d_out;

  char* ws = (char*)d_ws;
  size_t off = 0;
  auto carve = [&](size_t bytes) -> void* {
    size_t a = (off + 255) & ~(size_t)255;
    off = a + bytes;
    return (void*)(ws + a);
  };
  const size_t XE = (size_t)NTOK * DIMF;                       // 2,097,152
  const size_t WE = (size_t)DIMF * DIMF;                       // 262,144
  const size_t QE = (size_t)NB * NHEADS * NSEQ * HDIM;         // 2,097,152

  unsigned short* xhi  = (unsigned short*)carve(XE * 2);
  unsigned short* xlo  = (unsigned short*)carve(XE * 2);
  unsigned short* wqth = (unsigned short*)carve(WE * 2);
  unsigned short* wqtl = (unsigned short*)carve(WE * 2);
  unsigned short* wkth = (unsigned short*)carve(WE * 2);
  unsigned short* wktl = (unsigned short*)carve(WE * 2);
  unsigned short* qhi  = (unsigned short*)carve(QE * 2);
  unsigned short* qlo  = (unsigned short*)carve(QE * 2);
  unsigned short* kthi = (unsigned short*)carve(QE * 2);
  unsigned short* ktlo = (unsigned short*)carve(QE * 2);
  (void)ws_size; (void)in_sizes; (void)n_in; (void)out_size;

  k_split<<<dim3((unsigned)((XE + 255) / 256)), 256, 0, stream>>>(x, xhi, xlo, (int)XE);
  k_splitT<<<dim3((unsigned)(WE / 256)), 256, 0, stream>>>(WQ, wqth, wqtl);
  k_splitT<<<dim3((unsigned)(WE / 256)), 256, 0, stream>>>(WK, wkth, wktl);

  k_proj<<<dim3(NTOK / 16, DIMF / 16), 32, 0, stream>>>(xhi, xlo, wqth, wqtl, bQ, qhi, qlo, 0);
  k_proj<<<dim3(NTOK / 16, DIMF / 16), 32, 0, stream>>>(xhi, xlo, wkth, wktl, bK, kthi, ktlo, 1);

  const size_t ldsBytes = (size_t)16 * LSTR * sizeof(float);   // 131,328 B (<320KB/WGP)
  k_attn<<<dim3(NSEQ / 16, NB * NHEADS), 512, ldsBytes, stream>>>(qhi, qlo, kthi, ktlo, out);
}
